// FeedbackModelWeightedLoss_74311524155809
// MI455X (gfx1250) — matrix operand order repaired and checked by reading, not compiled, not run on hardware
//
#include <hip/hip_runtime.h>
#include <hip/hip_bf16.h>
#include <stdint.h>

#define B_   16
#define T_   1024
#define H_   1024
#define HH_  512
#define G4_  2048
#define S_   32
#define C_   3
#define NH_  16
#define DH_  64

typedef __attribute__((ext_vector_type(16))) __bf16 v16bf;
typedef __attribute__((ext_vector_type(8)))  float  v8f;

union AF { v16bf v; __bf16 e[16]; uint4 q[2]; };
union CF { v8f   v; float  e[8];  };

// Low 32 bits of a flat shared-memory address == LDS byte offset (LDS aperture).
__device__ __forceinline__ uint32_t lds_off_u32(const void* p) {
  return (uint32_t)(uintptr_t)p;
}

// Async global -> LDS copy, 16B per lane. Tracked by ASYNCcnt.
__device__ __forceinline__ void async_load_b128(uint32_t lds_off, const void* gptr) {
  uint64_t ga = (uint64_t)(uintptr_t)gptr;
  asm volatile("global_load_async_to_lds_b128 %0, %1, off"
               :: "v"(lds_off), "v"(ga) : "memory");
}
__device__ __forceinline__ void wait_async0() {
  asm volatile("s_wait_asynccnt 0" ::: "memory");
}

__device__ __forceinline__ float sigf(float x) { return 1.f / (1.f + __expf(-x)); }

// ---------------------------------------------------------------------------
// elementwise casts
// ---------------------------------------------------------------------------
__global__ void cast_f32_bf16(const float* __restrict__ src, __bf16* __restrict__ dst, int n) {
  int i = blockIdx.x * blockDim.x + threadIdx.x;
  int stride = gridDim.x * blockDim.x;
  for (; i < n; i += stride) dst[i] = (__bf16)src[i];
}

// src: rows(K) x cols(N) row-major f32 -> dst: cols(N) x rows(K) bf16
__global__ void transpose_cast_bf16(const float* __restrict__ src, __bf16* __restrict__ dst,
                                    int rows, int cols) {
  int i = blockIdx.x * blockDim.x + threadIdx.x;
  int n = rows * cols;
  int stride = gridDim.x * blockDim.x;
  for (; i < n; i += stride) {
    int k = i / cols, c = i % cols;
    dst[(size_t)c * rows + k] = (__bf16)src[i];
  }
}

__global__ void zero4(float* a) { if (threadIdx.x < 4) a[threadIdx.x] = 0.f; }

// ---------------------------------------------------------------------------
// bf16 WMMA GEMM: C[M,N] = A[M,K] @ Wt[N,K]^T + bias[N]
// Block = 256 threads (8 waves), block tile 64(M) x 128(N), K step 32.
// Double-buffered LDS stages fed by async loads (stage k+1 issued before
// computing stage k). All fragments are pulled into registers before the
// 4 back-to-back WMMAs so DS waits can be partial and overlap the XDL pipe.
// ---------------------------------------------------------------------------
__global__ __launch_bounds__(256)
void gemm_bf16_wmma(const __bf16* __restrict__ A, const __bf16* __restrict__ Wt,
                    const float* __restrict__ bias, float* __restrict__ Cout,
                    int M, int N, int K) {
  __shared__ __bf16 tA[2][64 * 32];
  __shared__ __bf16 tW[2][128 * 32];
  const int tid  = threadIdx.x;
  const int lane = tid & 31;
  const int wave = tid >> 5;
  const int wm = wave >> 1, wn = wave & 1;
  const int m0 = blockIdx.y * 64;
  const int n0 = blockIdx.x * 128;
  const int arow = tid >> 2, ach = tid & 3;      // A staging assignment
  const __bf16* gA = A + (size_t)(m0 + arow) * K + ach * 8;

  const v8f vzero = {0, 0, 0, 0, 0, 0, 0, 0};
  v8f acc0 = vzero, acc1 = vzero, acc2 = vzero, acc3 = vzero;
  const int nstage = K / 32;

  // prologue: stage 0 into buffer 0
  async_load_b128(lds_off_u32(&tA[0][arow * 32 + ach * 8]), gA);
  #pragma unroll
  for (int rep = 0; rep < 2; ++rep) {
    int c = tid + rep * 256;
    int row = c >> 2, ch = c & 3;
    async_load_b128(lds_off_u32(&tW[0][row * 32 + ch * 8]),
                    Wt + (size_t)(n0 + row) * K + ch * 8);
  }

  for (int si = 0; si < nstage; ++si) {
    const int cur = si & 1;
    wait_async0();       // this wave's stage-si loads landed
    __syncthreads();     // every wave's stage-si loads landed, compute(si-1) done

    if (si + 1 < nstage) {        // issue stage si+1 into the other buffer
      const int k1 = (si + 1) * 32;
      async_load_b128(lds_off_u32(&tA[cur ^ 1][arow * 32 + ach * 8]), gA + k1);
      #pragma unroll
      for (int rep = 0; rep < 2; ++rep) {
        int c = tid + rep * 256;
        int row = c >> 2, ch = c & 3;
        async_load_b128(lds_off_u32(&tW[cur ^ 1][row * 32 + ch * 8]),
                        Wt + (size_t)(n0 + row) * K + k1 + ch * 8);
      }
      if (si + 2 < nstage)  // pull stage si+2 toward L2
        __builtin_prefetch(gA + (si + 2) * 32, 0, 1);
    }

    // pull ALL fragments into registers first (partial DS waits can overlap
    // the WMMA pipe), then issue 4 WMMAs back-to-back.
    AF a;
    {
      int row = wm * 16 + (lane & 15);
      int base = (lane < 16) ? 0 : 8;
      a.q[0] = *(const uint4*)&tA[cur][row * 32 + base];
      a.q[1] = *(const uint4*)&tA[cur][row * 32 + base + 16];
    }
    AF b[4];
    #pragma unroll
    for (int nt = 0; nt < 4; ++nt) {
      int row  = wn * 64 + nt * 16 + (lane & 15);
      int koff = (lane < 16) ? 0 : 8;
      b[nt].q[0] = *(const uint4*)&tW[cur][row * 32 + koff];
      b[nt].q[1] = *(const uint4*)&tW[cur][row * 32 + koff + 16];
    }
    acc0 = __builtin_amdgcn_wmma_f32_16x16x32_bf16(false, a.v, false, b[0].v,
                                                   (short)0, acc0, false, false);
    acc1 = __builtin_amdgcn_wmma_f32_16x16x32_bf16(false, a.v, false, b[1].v,
                                                   (short)0, acc1, false, false);
    acc2 = __builtin_amdgcn_wmma_f32_16x16x32_bf16(false, a.v, false, b[2].v,
                                                   (short)0, acc2, false, false);
    acc3 = __builtin_amdgcn_wmma_f32_16x16x32_bf16(false, a.v, false, b[3].v,
                                                   (short)0, acc3, false, false);
  }

  #pragma unroll
  for (int nt = 0; nt < 4; ++nt) {
    CF c; c.v = (nt == 0) ? acc0 : (nt == 1) ? acc1 : (nt == 2) ? acc2 : acc3;
    int n = n0 + wn * 64 + nt * 16 + (lane & 15);
    float bv = bias ? bias[n] : 0.f;
    #pragma unroll
    for (int r = 0; r < 8; ++r) {
      int m = m0 + wm * 16 + r + ((lane < 16) ? 0 : 8);
      Cout[(size_t)m * N + n] = c.e[r] + bv;
    }
  }
}

// ---------------------------------------------------------------------------
// Bidirectional LSTM scan. grid = 2 (direction), block = 512 (16 waves).
// B=16 == WMMA M tile. Wave w owns hidden slice j in [w*32, w*32+32):
// computes all 4 gates for its slice (8 WMMA n-tiles), keeps cell state c
// in accumulator-layout VGPRs, publishes h as bf16 in LDS for all waves.
// Per k-iter: A fragment + all 8 Whh B fragments are loaded first, then the
// 8 WMMAs issue back-to-back; xs gate loads are hoisted above the WMMA loop.
// ---------------------------------------------------------------------------
__global__ __launch_bounds__(512)
void lstm_scan(const float* __restrict__ xs_f, const float* __restrict__ xs_b,
               const __bf16* __restrict__ whh_f, const __bf16* __restrict__ whh_b,
               float* __restrict__ enc) {
  __shared__ __bf16 hbuf[16 * HH_];
  const int dir = blockIdx.x;
  const float*  xs  = dir ? xs_b : xs_f;
  const __bf16* whh = dir ? whh_b : whh_f;
  const int tid = threadIdx.x, lane = tid & 31, wave = tid >> 5;
  const int jbase = wave * 32;
  const v8f vzero = {0, 0, 0, 0, 0, 0, 0, 0};

  for (int i = tid; i < 16 * HH_; i += blockDim.x) hbuf[i] = (__bf16)0.f;
  __syncthreads();

  CF cst[2];
  cst[0].v = vzero; cst[1].v = vzero;

  for (int step = 0; step < T_; ++step) {
    const int t = dir ? (T_ - 1 - step) : step;

    // hoisted xs gate loads: independent of h, overlap with WMMA below
    float xsv[4][2][8];
    #pragma unroll
    for (int g = 0; g < 4; ++g)
      #pragma unroll
      for (int tt = 0; tt < 2; ++tt)
        #pragma unroll
        for (int r = 0; r < 8; ++r) {
          int m = r + ((lane < 16) ? 0 : 8);
          int j = jbase + tt * 16 + (lane & 15);
          xsv[g][tt][r] = xs[((size_t)m * T_ + t) * G4_ + g * HH_ + j];
        }

    v8f acc[8];
    #pragma unroll
    for (int q = 0; q < 8; ++q) acc[q] = vzero;

    for (int k0 = 0; k0 < HH_; k0 += 32) {
      AF a;
      {
        int m = lane & 15;
        int base = (lane < 16) ? 0 : 8;
        a.q[0] = *(const uint4*)&hbuf[m * HH_ + k0 + base];
        a.q[1] = *(const uint4*)&hbuf[m * HH_ + k0 + base + 16];
      }
      AF b[8];
      #pragma unroll
      for (int g = 0; g < 4; ++g) {
        #pragma unroll
        for (int tt = 0; tt < 2; ++tt) {
          int n = g * HH_ + jbase + tt * 16 + (lane & 15);
          int koff = (lane < 16) ? 0 : 8;
          const uint4* p = (const uint4*)(whh + (size_t)n * HH_ + k0 + koff);
          b[g * 2 + tt].q[0] = p[0];
          b[g * 2 + tt].q[1] = p[2];
        }
      }
      #pragma unroll
      for (int q = 0; q < 8; ++q)
        acc[q] = __builtin_amdgcn_wmma_f32_16x16x32_bf16(
            false, a.v, false, b[q].v, (short)0, acc[q], false, false);
    }

    // prefetch next step's xs slice for this wave
    if (step + 1 < T_) {
      int tn = dir ? (T_ - 2 - step) : step + 1;
      __builtin_prefetch(xs + ((size_t)(lane & 15) * T_ + tn) * G4_ + jbase, 0, 1);
    }

    __syncthreads(); // all waves finished reading old h

    #pragma unroll
    for (int tt = 0; tt < 2; ++tt) {
      CF gi, gf, gg, go;
      gi.v = acc[0 * 2 + tt]; gf.v = acc[1 * 2 + tt];
      gg.v = acc[2 * 2 + tt]; go.v = acc[3 * 2 + tt];
      #pragma unroll
      for (int r = 0; r < 8; ++r) {
        int m = r + ((lane < 16) ? 0 : 8);
        int j = jbase + tt * 16 + (lane & 15);
        float iv = gi.e[r] + xsv[0][tt][r];
        float fv = gf.e[r] + xsv[1][tt][r];
        float gv = gg.e[r] + xsv[2][tt][r];
        float ov = go.e[r] + xsv[3][tt][r];
        float c = sigf(fv) * cst[tt].e[r] + sigf(iv) * tanhf(gv);
        cst[tt].e[r] = c;
        float h = sigf(ov) * tanhf(c);
        hbuf[m * HH_ + j] = (__bf16)h;
        enc[((size_t)m * T_ + t) * H_ + dir * HH_ + j] = h;
      }
    }
    __syncthreads(); // h published for next step
  }
}

// ---------------------------------------------------------------------------
// span mean-pool + layernorm. grid = B*S, block = 256 (4 dims/thread).
// ---------------------------------------------------------------------------
__global__ __launch_bounds__(256)
void span_pool_ln(const float* __restrict__ enc, const int* __restrict__ heads,
                  const int* __restrict__ tails, const float* __restrict__ g,
                  const float* __restrict__ bb, float* __restrict__ feat) {
  __shared__ float red[256];
  const int bs = blockIdx.x;
  const int b = bs / S_;
  const int tid = threadIdx.x;
  const int head = heads[bs], tail = tails[bs];

  float acc[4] = {0, 0, 0, 0};
  int cnt = 0;
  for (int t = head + 1; t < tail; ++t) {
    const float* row = enc + ((size_t)b * T_ + t) * H_;
    #pragma unroll
    for (int u = 0; u < 4; ++u) acc[u] += row[tid + u * 256];
    ++cnt;
  }
  float inv = 1.f / fmaxf((float)cnt, 1.f);
  float x[4];
  #pragma unroll
  for (int u = 0; u < 4; ++u) x[u] = acc[u] * inv;

  float lsum = x[0] + x[1] + x[2] + x[3];
  red[tid] = lsum; __syncthreads();
  for (int off = 128; off; off >>= 1) { if (tid < off) red[tid] += red[tid + off]; __syncthreads(); }
  float mean = red[0] / (float)H_;
  __syncthreads();
  float lv = 0;
  #pragma unroll
  for (int u = 0; u < 4; ++u) { float d = x[u] - mean; lv += d * d; }
  red[tid] = lv; __syncthreads();
  for (int off = 128; off; off >>= 1) { if (tid < off) red[tid] += red[tid + off]; __syncthreads(); }
  float rstd = rsqrtf(red[0] / (float)H_ + 1e-7f);
  #pragma unroll
  for (int u = 0; u < 4; ++u) {
    int i = tid + u * 256;
    feat[(size_t)bs * H_ + i] = (x[u] - mean) * rstd * g[i] + bb[i];
  }
}

// ---------------------------------------------------------------------------
// attention over S=32 spans: one wave per (b, head); K/V staged in LDS.
// ---------------------------------------------------------------------------
__global__ __launch_bounds__(32)
void attention(const float* __restrict__ q, const float* __restrict__ k,
               const float* __restrict__ v, const int* __restrict__ am,
               float* __restrict__ ctx) {
  __shared__ float kb[S_ * DH_];
  __shared__ float vb[S_ * DH_];
  const int h = blockIdx.x, b = blockIdx.y;
  const int lane = threadIdx.x;
  for (int i = lane; i < S_ * DH_; i += 32) {
    int s = i / DH_, d = i % DH_;
    size_t off = ((size_t)(b * S_ + s)) * H_ + h * DH_ + d;
    kb[i] = k[off]; vb[i] = v[off];
  }
  __syncthreads();

  float qr[DH_];
  size_t qoff = ((size_t)(b * S_ + lane)) * H_ + h * DH_;
  #pragma unroll
  for (int d = 0; d < DH_; ++d) qr[d] = q[qoff + d];

  const int mq = am[b * S_ + lane];
  float sc[S_];
  float mx = -1e30f;
  for (int j = 0; j < S_; ++j) {
    float dot = 0.f;
    #pragma unroll
    for (int d = 0; d < DH_; ++d) dot += qr[d] * kb[j * DH_ + d];
    dot *= 0.125f; // 1/sqrt(64)
    if (!(mq && am[b * S_ + j])) dot = -1e9f;
    sc[j] = dot; mx = fmaxf(mx, dot);
  }
  float sum = 0.f;
  for (int j = 0; j < S_; ++j) { sc[j] = __expf(sc[j] - mx); sum += sc[j]; }
  float inv = 1.f / sum;
  #pragma unroll
  for (int d = 0; d < DH_; ++d) {
    float o = 0.f;
    for (int j = 0; j < S_; ++j) o += sc[j] * vb[j * DH_ + d];
    ctx[qoff + d] = o * inv;
  }
}

// ---------------------------------------------------------------------------
// feat2 = layernorm(wo_out + feat)  (wo_out already includes bo via gemm bias)
// ---------------------------------------------------------------------------
__global__ __launch_bounds__(256)
void residual_ln(const float* __restrict__ wo_out, const float* __restrict__ feat,
                 const float* __restrict__ g, const float* __restrict__ bb,
                 float* __restrict__ feat2) {
  __shared__ float red[256];
  const int bs = blockIdx.x, tid = threadIdx.x;
  float x[4];
  #pragma unroll
  for (int u = 0; u < 4; ++u) {
    int i = tid + u * 256;
    x[u] = wo_out[(size_t)bs * H_ + i] + feat[(size_t)bs * H_ + i];
  }
  float lsum = x[0] + x[1] + x[2] + x[3];
  red[tid] = lsum; __syncthreads();
  for (int off = 128; off; off >>= 1) { if (tid < off) red[tid] += red[tid + off]; __syncthreads(); }
  float mean = red[0] / (float)H_;
  __syncthreads();
  float lv = 0;
  #pragma unroll
  for (int u = 0; u < 4; ++u) { float d = x[u] - mean; lv += d * d; }
  red[tid] = lv; __syncthreads();
  for (int off = 128; off; off >>= 1) { if (tid < off) red[tid] += red[tid + off]; __syncthreads(); }
  float rstd = rsqrtf(red[0] / (float)H_ + 1e-7f);
  #pragma unroll
  for (int u = 0; u < 4; ++u) {
    int i = tid + u * 256;
    feat2[(size_t)bs * H_ + i] = (x[u] - mean) * rstd * g[i] + bb[i];
  }
}

// ---------------------------------------------------------------------------
// logits + CE/focal loss accumulation. grid = B*S.
// accum: [ce_num, ce_den, focal_num, focal_den]
// ---------------------------------------------------------------------------
__global__ __launch_bounds__(256)
void loss_kernel(const float* __restrict__ feat2, const float* __restrict__ Wc,
                 const float* __restrict__ bc, const int* __restrict__ labels,
                 const float* __restrict__ wf, const int* __restrict__ am,
                 float* __restrict__ accum) {
  __shared__ float r0[256], r1[256], r2[256];
  const int bs = blockIdx.x, tid = threadIdx.x;
  float p0 = 0, p1 = 0, p2 = 0;
  for (int kk = tid; kk < H_; kk += 256) {
    float x = feat2[(size_t)bs * H_ + kk];
    p0 += x * Wc[kk * 3 + 0];
    p1 += x * Wc[kk * 3 + 1];
    p2 += x * Wc[kk * 3 + 2];
  }
  r0[tid] = p0; r1[tid] = p1; r2[tid] = p2; __syncthreads();
  for (int off = 128; off; off >>= 1) {
    if (tid < off) { r0[tid] += r0[tid + off]; r1[tid] += r1[tid + off]; r2[tid] += r2[tid + off]; }
    __syncthreads();
  }
  if (tid == 0) {
    float l0 = r0[0] + bc[0], l1 = r1[0] + bc[1], l2 = r2[0] + bc[2];
    float mx = fmaxf(l0, fmaxf(l1, l2));
    float e0 = __expf(l0 - mx), e1 = __expf(l1 - mx), e2 = __expf(l2 - mx);
    float lse = mx + __logf(e0 + e1 + e2);
    float lp0 = l0 - lse, lp1 = l1 - lse, lp2 = l2 - lse;
    int lab = labels[bs];
    int valid = (lab >= 0) ? 1 : 0;
    int lc = lab < 0 ? 0 : (lab > 2 ? 2 : lab);
    float lp = (lc == 0) ? lp0 : (lc == 1) ? lp1 : lp2;
    float mask = (float)am[bs];
    float spanm = mask * (float)valid;
    float ce = -lp * wf[bs];
    float pt = __expf(lp);
    float om = 1.f - pt;
    float ft = -(om * om * lp);
    atomicAdd(&accum[0], ce * spanm);
    atomicAdd(&accum[1], spanm);
    atomicAdd(&accum[2], ft * (float)valid);
    atomicAdd(&accum[3], (float)valid);
  }
}

__global__ void finalize(const float* __restrict__ accum, float* __restrict__ out) {
  out[0] = accum[0] / fmaxf(accum[1], 1.f) + accum[2] / fmaxf(accum[3], 1.f);
}

// ---------------------------------------------------------------------------
extern "C" void kernel_launch(void* const* d_in, const int* in_sizes, int n_in,
                              void* d_out, int out_size, void* d_ws, size_t ws_size,
                              hipStream_t stream) {
  const float* hidden = (const float*)d_in[0];
  const int*   heads  = (const int*)d_in[1];
  const int*   tails  = (const int*)d_in[2];
  const int*   amask  = (const int*)d_in[3];
  const int*   labels = (const int*)d_in[4];
  const float* wfac   = (const float*)d_in[5];
  const float* Wih_f  = (const float*)d_in[6];
  const float* Whh_f  = (const float*)d_in[7];
  const float* b_f    = (const float*)d_in[8];
  const float* Wih_b  = (const float*)d_in[9];
  const float* Whh_b  = (const float*)d_in[10];
  const float* b_b    = (const float*)d_in[11];
  const float* ln1g   = (const float*)d_in[12];
  const float* ln1b   = (const float*)d_in[13];
  const float* Wq     = (const float*)d_in[14];
  const float* bq     = (const float*)d_in[15];
  const float* Wk     = (const float*)d_in[16];
  const float* bk     = (const float*)d_in[17];
  const float* Wv     = (const float*)d_in[18];
  const float* bv     = (const float*)d_in[19];
  const float* Wo     = (const float*)d_in[20];
  const float* bo     = (const float*)d_in[21];
  const float* ln2g   = (const float*)d_in[22];
  const float* ln2b   = (const float*)d_in[23];
  const float* Wc     = (const float*)d_in[24];
  const float* bc     = (const float*)d_in[25];

  // workspace layout (256B aligned slabs)
  char* base = (char*)d_ws;
  size_t off = 0;
  auto alloc = [&](size_t bytes) { char* p = base + off; off += (bytes + 255) & ~(size_t)255; return p; };
  __bf16* xbf    = (__bf16*)alloc((size_t)B_ * T_ * H_ * 2);
  __bf16* wihf   = (__bf16*)alloc((size_t)G4_ * H_ * 2);
  __bf16* wihb   = (__bf16*)alloc((size_t)G4_ * H_ * 2);
  __bf16* whhf   = (__bf16*)alloc((size_t)G4_ * HH_ * 2);
  __bf16* whhb   = (__bf16*)alloc((size_t)G4_ * HH_ * 2);
  __bf16* wqt    = (__bf16*)alloc((size_t)H_ * H_ * 2);
  __bf16* wkt    = (__bf16*)alloc((size_t)H_ * H_ * 2);
  __bf16* wvt    = (__bf16*)alloc((size_t)H_ * H_ * 2);
  __bf16* wot    = (__bf16*)alloc((size_t)H_ * H_ * 2);
  float*  xsf    = (float*)alloc((size_t)B_ * T_ * G4_ * 4);
  float*  xsb    = (float*)alloc((size_t)B_ * T_ * G4_ * 4);
  float*  enc    = (float*)alloc((size_t)B_ * T_ * H_ * 4);
  float*  feat   = (float*)alloc((size_t)B_ * S_ * H_ * 4);
  __bf16* featbf = (__bf16*)alloc((size_t)B_ * S_ * H_ * 2);
  float*  qf     = (float*)alloc((size_t)B_ * S_ * H_ * 4);
  float*  kf     = (float*)alloc((size_t)B_ * S_ * H_ * 4);
  float*  vf     = (float*)alloc((size_t)B_ * S_ * H_ * 4);
  float*  ctx    = (float*)alloc((size_t)B_ * S_ * H_ * 4);
  __bf16* ctxbf  = (__bf16*)alloc((size_t)B_ * S_ * H_ * 2);
  float*  wo_out = (float*)alloc((size_t)B_ * S_ * H_ * 4);
  float*  feat2  = (float*)alloc((size_t)B_ * S_ * H_ * 4);
  float*  accum  = (float*)alloc(64);
  (void)ws_size; (void)in_sizes; (void)n_in; (void)out_size;

  zero4<<<1, 32, 0, stream>>>(accum);

  cast_f32_bf16<<<2048, 256, 0, stream>>>(hidden, xbf, B_ * T_ * H_);
  cast_f32_bf16<<<1024, 256, 0, stream>>>(Wih_f, wihf, G4_ * H_);
  cast_f32_bf16<<<1024, 256, 0, stream>>>(Wih_b, wihb, G4_ * H_);
  cast_f32_bf16<<<512, 256, 0, stream>>>(Whh_f, whhf, G4_ * HH_);
  cast_f32_bf16<<<512, 256, 0, stream>>>(Whh_b, whhb, G4_ * HH_);
  transpose_cast_bf16<<<1024, 256, 0, stream>>>(Wq, wqt, H_, H_);
  transpose_cast_bf16<<<1024, 256, 0, stream>>>(Wk, wkt, H_, H_);
  transpose_cast_bf16<<<1024, 256, 0, stream>>>(Wv, wvt, H_, H_);
  transpose_cast_bf16<<<1024, 256, 0, stream>>>(Wo, wot, H_, H_);

  // xs = x @ Wih^T + b   (M=16384, N=2048, K=1024), both directions
  gemm_bf16_wmma<<<dim3(G4_ / 128, (B_ * T_) / 64), 256, 0, stream>>>(
      xbf, wihf, b_f, xsf, B_ * T_, G4_, H_);
  gemm_bf16_wmma<<<dim3(G4_ / 128, (B_ * T_) / 64), 256, 0, stream>>>(
      xbf, wihb, b_b, xsb, B_ * T_, G4_, H_);

  lstm_scan<<<2, 512, 0, stream>>>(xsf, xsb, whhf, whhb, enc);

  span_pool_ln<<<B_ * S_, 256, 0, stream>>>(enc, heads, tails, ln1g, ln1b, feat);
  cast_f32_bf16<<<256, 256, 0, stream>>>(feat, featbf, B_ * S_ * H_);

  gemm_bf16_wmma<<<dim3(H_ / 128, (B_ * S_) / 64), 256, 0, stream>>>(
      featbf, wqt, bq, qf, B_ * S_, H_, H_);
  gemm_bf16_wmma<<<dim3(H_ / 128, (B_ * S_) / 64), 256, 0, stream>>>(
      featbf, wkt, bk, kf, B_ * S_, H_, H_);
  gemm_bf16_wmma<<<dim3(H_ / 128, (B_ * S_) / 64), 256, 0, stream>>>(
      featbf, wvt, bv, vf, B_ * S_, H_, H_);

  attention<<<dim3(NH_, B_), 32, 0, stream>>>(qf, kf, vf, amask, ctx);
  cast_f32_bf16<<<256, 256, 0, stream>>>(ctx, ctxbf, B_ * S_ * H_);

  gemm_bf16_wmma<<<dim3(H_ / 128, (B_ * S_) / 64), 256, 0, stream>>>(
      ctxbf, wot, bo, wo_out, B_ * S_, H_, H_);

  residual_ln<<<B_ * S_, 256, 0, stream>>>(wo_out, feat, ln2g, ln2b, feat2);
  loss_kernel<<<B_ * S_, 256, 0, stream>>>(feat2, Wc, bc, labels, wfac, amask, accum);
  finalize<<<1, 1, 0, stream>>>(accum, (float*)d_out);
}
